// ProcessorCNN_54073638257192
// MI455X (gfx1250) — matrix operand order repaired and checked by reading, not compiled, not run on hardware
//
#include <hip/hip_runtime.h>
#include <hip/hip_bf16.h>

#define D        256
#define STRIPS   3
#define ROWS     48                 // 3 WMMA row strips per block
#define ASTRIDE  (2*D + 8)          // 520 bf16 per LDS row: [x(256) | mean(256)] + pad
#define HSTRIDE  (D + 4)            // 260 f32 per LDS row (aliases A buffer exactly)
#define LN_EPS   1e-5f
#define KTILES   16                 // K = 512 (x | mean) in tiles of 32
// packed B: Bp[kt][g][col][e], e=0..15 contiguous bf16 (32 B per lane-fragment)
#define BPACK_PER_LAYER (KTILES * 2 * D * 16)   // 131072 bf16 = 256 KiB

typedef __attribute__((ext_vector_type(16))) __bf16 v16bf;
typedef __attribute__((ext_vector_type(8)))  __bf16 v8bf;
typedef __attribute__((ext_vector_type(8)))  float  v8f;

// ---- pre-pack [Wself;Wneigh] (both layers) into WMMA B-fragment order, bf16 ----
__global__ __launch_bounds__(256) void pack_weights_kernel(
    const float* __restrict__ wself,    // [2,256,256]
    const float* __restrict__ wneigh,   // [2,256,256]
    __bf16*      __restrict__ bp)       // [2][16][2][256][16]
{
    int tid = blockIdx.x * 256 + threadIdx.x;     // 0 .. 2*131072-1
    int e   = tid & 15;
    int col = (tid >> 4) & (D - 1);
    int g   = (tid >> 12) & 1;
    int kt  = (tid >> 13) & (KTILES - 1);
    int l   = tid >> 17;
    int K   = kt * 32 + g * 16 + e;               // 0..511 over [x|mean]
    const float* src = (K < D) ? (wself  + (size_t)l * D * D + (size_t)K * D + col)
                               : (wneigh + (size_t)l * D * D + (size_t)(K - D) * D + col);
    bp[tid] = (__bf16)(*src);
}

__global__ __launch_bounds__(256) void gnn_layer_kernel(
    const float*  __restrict__ xin,     // [n, 256]
    const int*    __restrict__ nbr,     // [n, 6]
    const __bf16* __restrict__ bpack,   // [16][2][256][16] packed B fragments
    const float*  __restrict__ bias,    // [256]
    const float*  __restrict__ lnsc,    // [256]
    const float*  __restrict__ lnof,    // [256]
    float*        __restrict__ out,     // [n, 256]
    int n)
{
    // 49920 B, dual-purpose: bf16 A-matrix [48][520] then f32 H [48][260]
    __shared__ __align__(16) unsigned char smem[ROWS * ASTRIDE * 2];
    __shared__ float2 partials[ROWS * 16];
    __shared__ float2 stats[ROWS];

    __bf16* Albs = (__bf16*)smem;
    float*  Hlds = (float*)smem;

    const int t    = threadIdx.x;
    const int lane = t & 31;
    const int wv   = t >> 5;            // wave id 0..7 -> 32-column slab
    const int row0 = blockIdx.x * ROWS;

    // ---------------- stage-in: x rows + 6-neighbor mean -> bf16 LDS ----------------
    #pragma unroll
    for (int i = 0; i < 3; ++i) {
        const int c    = t + i * 256;        // 0..767
        const int r    = c >> 4;             // 0..47
        const int cb   = (c & 15) * 16;      // column base
        const int grow = row0 + r;
        float xv[16], mv[16];
        #pragma unroll
        for (int k = 0; k < 16; ++k) { xv[k] = 0.f; mv[k] = 0.f; }
        if (grow < n) {
            const float4* xr = (const float4*)(xin + (size_t)grow * D + cb);
            #pragma unroll
            for (int q = 0; q < 4; ++q) {
                float4 v = xr[q];
                xv[q*4+0] = v.x; xv[q*4+1] = v.y; xv[q*4+2] = v.z; xv[q*4+3] = v.w;
            }
            const int* nb = nbr + (size_t)grow * 6;
            for (int j = 0; j < 6; ++j) {
                const float4* nr = (const float4*)(xin + (size_t)nb[j] * D + cb);
                #pragma unroll
                for (int q = 0; q < 4; ++q) {
                    float4 v = nr[q];
                    mv[q*4+0] += v.x; mv[q*4+1] += v.y; mv[q*4+2] += v.z; mv[q*4+3] += v.w;
                }
            }
        }
        __bf16* ax = Albs + (size_t)r * ASTRIDE + cb;       // K = 0..255  : x
        __bf16* am = Albs + (size_t)r * ASTRIDE + D + cb;   // K = 256..511: mean
        const float inv6 = 1.f / 6.f;
        #pragma unroll
        for (int k = 0; k < 16; ++k) {
            ax[k] = (__bf16)xv[k];
            am[k] = (__bf16)(mv[k] * inv6);
        }
    }
    __syncthreads();

    // ---------------- K-loop: h = [x|m] @ [[Wself];[Wneigh]] via WMMA bf16 ----------------
    v8f acc[STRIPS][2];
    #pragma unroll
    for (int s = 0; s < STRIPS; ++s)
        #pragma unroll
        for (int j = 0; j < 2; ++j) {
            v8f z = {0.f,0.f,0.f,0.f,0.f,0.f,0.f,0.f};
            acc[s][j] = z;
        }

    const int cbase = wv * 32;                  // this wave's 32-column slab
    const int hlf   = lane & 15;
    const int kbA   = (lane < 16) ? 0 : 16;     // A fragment K sub-offset
    const int gB    = (lane < 16) ? 0 : 1;      // B fragment K group

    // lane's packed-B base: Bp[0][gB][cbase+hlf][0]; per kt advance 2*256*16 elems
    const __bf16* bp = bpack + ((size_t)gB * D + cbase + hlf) * 16;

    for (int kt = 0; kt < KTILES; ++kt) {
        // A fragments (one per row-strip) from padded bf16 LDS; bank-balanced B128 reads
        v16bf afrag[STRIPS];
        #pragma unroll
        for (int s = 0; s < STRIPS; ++s) {
            const __bf16* ap = Albs + (size_t)(s * 16 + hlf) * ASTRIDE + kt * 32 + kbA;
            v8bf lo = *(const v8bf*)ap;         // K = kt*32 + kbA + 0..7
            v8bf hi = *(const v8bf*)(ap + 8);   // K = kt*32 + kbA + 8..15
            afrag[s] = __builtin_shufflevector(lo, hi,
                        0,1,2,3,4,5,6,7,8,9,10,11,12,13,14,15);
        }
        #pragma unroll
        for (int j = 0; j < 2; ++j) {
            // contiguous 32 B per lane; lanes 0-15 cover 512 B contiguous
            v16bf bfrag = *(const v16bf*)(bp + (size_t)j * 16 * 16);
            #pragma unroll
            for (int s = 0; s < STRIPS; ++s)
                acc[s][j] = __builtin_amdgcn_wmma_f32_16x16x32_bf16(
                    false, afrag[s], false, bfrag, (short)0, acc[s][j],
                    false, false);
        }
        bp += (size_t)2 * D * 16;               // next kt
    }

    // ---------------- spill accumulators to LDS (alias A buffer) ----------------
    __syncthreads();   // everyone done reading A
    const int rsub = (lane < 16) ? 0 : 8;
    #pragma unroll
    for (int s = 0; s < STRIPS; ++s)
        #pragma unroll
        for (int j = 0; j < 2; ++j) {
            const int col = cbase + j * 16 + hlf;
            #pragma unroll
            for (int v = 0; v < 8; ++v)
                Hlds[(size_t)(s * 16 + rsub + v) * HSTRIDE + col] = acc[s][j][v];
        }
    __syncthreads();

    // ---------------- epilogue: +x +bias, LayerNorm, ReLU ----------------
    float hreg[3][16];
    #pragma unroll
    for (int i = 0; i < 3; ++i) {
        const int c    = t + i * 256;
        const int r    = c >> 4;
        const int cb   = (c & 15) * 16;
        const int grow = row0 + r;
        float s1 = 0.f, s2 = 0.f;
        if (grow < n) {
            const float* xr = xin + (size_t)grow * D + cb;
            const float* hp = Hlds + (size_t)r * HSTRIDE + cb;
            #pragma unroll
            for (int k = 0; k < 16; ++k) {
                float v = hp[k] + xr[k] + bias[cb + k];
                hreg[i][k] = v;
                s1 += v;
                s2 += v * v;
            }
        }
        partials[r * 16 + (c & 15)] = make_float2(s1, s2);
    }
    __syncthreads();
    if (t < ROWS) {
        float s1 = 0.f, s2 = 0.f;
        #pragma unroll
        for (int k = 0; k < 16; ++k) {
            float2 p = partials[t * 16 + k];
            s1 += p.x; s2 += p.y;
        }
        const float mean = s1 * (1.f / D);
        const float var  = s2 * (1.f / D) - mean * mean;
        stats[t] = make_float2(mean, rsqrtf(var + LN_EPS));
    }
    __syncthreads();
    #pragma unroll
    for (int i = 0; i < 3; ++i) {
        const int c    = t + i * 256;
        const int r    = c >> 4;
        const int cb   = (c & 15) * 16;
        const int grow = row0 + r;
        if (grow < n) {
            const float2 st = stats[r];
            float o[16];
            #pragma unroll
            for (int k = 0; k < 16; ++k) {
                float v = (hreg[i][k] - st.x) * st.y * lnsc[cb + k] + lnof[cb + k];
                o[k] = fmaxf(v, 0.f);
            }
            float4* op = (float4*)(out + (size_t)grow * D + cb);
            #pragma unroll
            for (int q = 0; q < 4; ++q)
                op[q] = make_float4(o[q*4+0], o[q*4+1], o[q*4+2], o[q*4+3]);
        }
    }
}

extern "C" void kernel_launch(void* const* d_in, const int* in_sizes, int n_in,
                              void* d_out, int out_size, void* d_ws, size_t ws_size,
                              hipStream_t stream) {
    const float* nodes  = (const float*)d_in[0];   // [N,256] f32
    const int*   nbr    = (const int*)  d_in[1];   // [N,6]   i32
    const float* wself  = (const float*)d_in[2];   // [2,256,256]
    const float* wneigh = (const float*)d_in[3];   // [2,256,256]
    const float* bneigh = (const float*)d_in[4];   // [2,256]
    const float* lnsc   = (const float*)d_in[5];   // [2,256]
    const float* lnof   = (const float*)d_in[6];   // [2,256]
    float*       outp   = (float*)d_out;

    const int n = in_sizes[0] / D;                 // 262144

    // workspace layout: [ xmid : n*256 f32 ][ Bpack : 2 layers * 256 KiB bf16 ]
    float*  xmid  = (float*)d_ws;
    __bf16* bpack = (__bf16*)((char*)d_ws + (size_t)n * D * sizeof(float));

    dim3 block(256);

    // pack both layers' weights into bf16 WMMA B-fragment order (tiny, L2-resident)
    pack_weights_kernel<<<dim3(2 * BPACK_PER_LAYER / 256), block, 0, stream>>>(
        wself, wneigh, bpack);

    dim3 grid((n + ROWS - 1) / ROWS);
    // layer 0: nodes -> ws  (not in-place: gather reads arbitrary rows)
    gnn_layer_kernel<<<grid, block, 0, stream>>>(
        nodes, nbr, bpack, bneigh, lnsc, lnof, xmid, n);
    // layer 1: ws -> out
    gnn_layer_kernel<<<grid, block, 0, stream>>>(
        xmid, nbr, bpack + BPACK_PER_LAYER,
        bneigh + D, lnsc + D, lnof + D, outp, n);
}
